// RNN_9543417331952
// MI455X (gfx1250) — matrix operand order repaired and checked by reading, not compiled, not run on hardware
//
#include <hip/hip_runtime.h>
#include <hip/hip_bf16.h>

// Fused LSTM for MI455X (gfx1250, wave32, v_wmma_f32_16x16x32_f16).
// B=4096, T=256, I=128, H=64, G=4H=256, O=10.
// One workgroup (4 wave32s, 128 threads) owns 16 batch rows for the entire
// time scan -> 256 workgroups to fill a large WGP array.
// Wave w owns N-tiles {w, w+4, w+8, w+12}: the SAME 16-column slice of each
// gate i/f/g/o, so the cell update and cell state c live entirely in
// registers. Only h round-trips through LDS (redistributed along K for the
// next step's WMMA). Single barrier per timestep; x(t+1) prefetched during
// step t's WMMAs via double-buffered LDS tiles.

typedef __attribute__((ext_vector_type(16))) _Float16 v16h;
typedef __attribute__((ext_vector_type(8)))  _Float16 v8h;
typedef __attribute__((ext_vector_type(8)))  float    v8f;

constexpr int Bv = 4096;
constexpr int Tv = 256;
constexpr int Iv = 128;
constexpr int Hv = 64;
constexpr int Gv = 256;   // 4H
constexpr int Ov = 10;
constexpr int BT = 16;    // batch tile per workgroup
constexpr int NT = 128;   // threads per block (4 wave32s)

struct __align__(16) SMem {
    _Float16 wihF[4 * 16 * 32 * 16];  // 64KB: W_ih^T B-fragments [kk][ntile][lane][h]
    _Float16 whhF[2 * 16 * 32 * 16];  // 32KB: W_hh^T B-fragments
    _Float16 sX[2][BT * Iv];          // 2 x 4KB: double-buffered x tile (f16)
    _Float16 sH[2][BT * Hv];          // 2 x 2KB: double-buffered h state (f16)
    float    bias[Gv];                // 1KB: b_ih + b_hh
};

__device__ __forceinline__ float sigm_fast(float x) {
    return 1.0f / (1.0f + __expf(-x));
}
__device__ __forceinline__ float tanh_fast(float x) {
    float e = __expf(2.0f * x);       // 1 - 2/(e^{2x}+1): saturates correctly
    return 1.0f - 2.0f / (e + 1.0f);
}

__device__ __forceinline__ v16h load_a_frag(const _Float16* p) {
    // Documented 16-bit A layout: lanes 0-15 hold K {0-7,16-23}, lanes 16-31
    // hold K {8-15,24-31} -> two contiguous 16B LDS loads per lane.
    v8h lo = *(const v8h*)p;
    v8h hi = *(const v8h*)(p + 16);
    v16h a;
#pragma unroll
    for (int e = 0; e < 8; ++e) { a[e] = lo[e]; a[e + 8] = hi[e]; }
    return a;
}

__global__ __launch_bounds__(NT, 2)
void lstm_fused_kernel(const float* __restrict__ x,
                       const float* __restrict__ W_ih,
                       const float* __restrict__ W_hh,
                       const float* __restrict__ b_ih,
                       const float* __restrict__ b_hh,
                       const float* __restrict__ W_out,
                       const float* __restrict__ b_out,
                       float* __restrict__ out) {
    __shared__ SMem sm;

    const int tid  = threadIdx.x;
    const int lane = tid & 31;
    const int wave = tid >> 5;        // 0..3
    const int lh   = lane & 15;
    const int hi   = lane >> 4;
    const int b0   = blockIdx.x * BT;

    // ---- Setup: weight B-fragments (f16) in LDS ----
    // B-fragment element (h, lane) of tile (kk, nt): K = kk*32 + (h&7) + 8*hi + (h>>3)*16, N = nt*16 + (lane&15)
    for (int f = tid; f < 4 * 16 * 32 * 16; f += NT) {
        int h  = f & 15;
        int ln = (f >> 4) & 31;
        int nt = (f >> 9) & 15;
        int kk = f >> 13;
        int n  = nt * 16 + (ln & 15);
        int k  = kk * 32 + (h & 7) + ((ln >> 4) & 1) * 8 + (h >> 3) * 16;
        sm.wihF[f] = (_Float16)W_ih[n * Iv + k];
    }
    for (int f = tid; f < 2 * 16 * 32 * 16; f += NT) {
        int h  = f & 15;
        int ln = (f >> 4) & 31;
        int nt = (f >> 9) & 15;
        int kk = f >> 13;
        int n  = nt * 16 + (ln & 15);
        int k  = kk * 32 + (h & 7) + ((ln >> 4) & 1) * 8 + (h >> 3) * 16;
        sm.whhF[f] = (_Float16)W_hh[n * Hv + k];
    }
    for (int f = tid; f < Gv; f += NT)
        sm.bias[f] = b_ih[f] + b_hh[f];
    for (int f = tid; f < BT * Hv; f += NT)
        sm.sH[0][f] = (_Float16)0.0f;

    // ---- Stage x(0) into sX[0] ----
    const int xm = tid >> 3;           // row 0..15
    const int xs = (tid & 7) * 16;     // 16 floats per thread
    {
        const float4* xrow = (const float4*)(x + ((size_t)(b0 + xm) * Tv + 0) * Iv + xs);
        float4 v0 = xrow[0], v1 = xrow[1], v2 = xrow[2], v3 = xrow[3];
        v8h p0, p1;
        p0[0] = (_Float16)v0.x; p0[1] = (_Float16)v0.y;
        p0[2] = (_Float16)v0.z; p0[3] = (_Float16)v0.w;
        p0[4] = (_Float16)v1.x; p0[5] = (_Float16)v1.y;
        p0[6] = (_Float16)v1.z; p0[7] = (_Float16)v1.w;
        p1[0] = (_Float16)v2.x; p1[1] = (_Float16)v2.y;
        p1[2] = (_Float16)v2.z; p1[3] = (_Float16)v2.w;
        p1[4] = (_Float16)v3.x; p1[5] = (_Float16)v3.y;
        p1[6] = (_Float16)v3.z; p1[7] = (_Float16)v3.w;
        *(v8h*)&sm.sX[0][xm * Iv + xs]     = p0;
        *(v8h*)&sm.sX[0][xm * Iv + xs + 8] = p1;
    }
    __syncthreads();

    // ---- Per-wave ownership: single M-tile, column-tile = wave ----
    const int ct = wave;               // column-tile 0..3 within each gate
    const int cc = ct * 16 + lh;       // owned hidden column 0..63

    float bj[4];
#pragma unroll
    for (int j = 0; j < 4; ++j) bj[j] = sm.bias[j * Hv + cc];

    v8f creg;                          // cell state, register-resident
#pragma unroll
    for (int r = 0; r < 8; ++r) creg[r] = 0.0f;

    int cur = 0;
    for (int t = 0; t < Tv; ++t) {
        const int nxt = cur ^ 1;

        // ---- Issue global loads for x(t+1) (in flight during WMMAs) ----
        const int tn = (t + 1 < Tv) ? (t + 1) : t;
        const float4* xrow = (const float4*)(x + ((size_t)(b0 + xm) * Tv + tn) * Iv + xs);
        float4 xr0 = xrow[0], xr1 = xrow[1], xr2 = xrow[2], xr3 = xrow[3];

        // ---- Gates = bias + xTile @ W_ih^T + h @ W_hh^T ----
        v8f acc[4];
#pragma unroll
        for (int j = 0; j < 4; ++j)
#pragma unroll
            for (int r = 0; r < 8; ++r) acc[j][r] = bj[j];

#pragma unroll
        for (int kk = 0; kk < 4; ++kk) {           // K = 128 over x
            v16h a = load_a_frag(&sm.sX[cur][lh * Iv + kk * 32 + hi * 8]);
#pragma unroll
            for (int j = 0; j < 4; ++j) {
                const int nt = ct + 4 * j;         // gate j, column-tile ct
                v16h b = *(const v16h*)&sm.wihF[((kk * 16 + nt) * 32 + lane) * 16];
                acc[j] = __builtin_amdgcn_wmma_f32_16x16x32_f16(
                    false, a, false, b, (short)0, acc[j], false, false);
            }
        }
#pragma unroll
        for (int kk = 0; kk < 2; ++kk) {           // K = 64 over h
            v16h a = load_a_frag(&sm.sH[cur][lh * Hv + kk * 32 + hi * 8]);
#pragma unroll
            for (int j = 0; j < 4; ++j) {
                const int nt = ct + 4 * j;
                v16h b = *(const v16h*)&sm.whhF[((kk * 16 + nt) * 32 + lane) * 16];
                acc[j] = __builtin_amdgcn_wmma_f32_16x16x32_f16(
                    false, a, false, b, (short)0, acc[j], false, false);
            }
        }

        // ---- Park x(t+1) into the other buffer ----
        {
            v8h p0, p1;
            p0[0] = (_Float16)xr0.x; p0[1] = (_Float16)xr0.y;
            p0[2] = (_Float16)xr0.z; p0[3] = (_Float16)xr0.w;
            p0[4] = (_Float16)xr1.x; p0[5] = (_Float16)xr1.y;
            p0[6] = (_Float16)xr1.z; p0[7] = (_Float16)xr1.w;
            p1[0] = (_Float16)xr2.x; p1[1] = (_Float16)xr2.y;
            p1[2] = (_Float16)xr2.z; p1[3] = (_Float16)xr2.w;
            p1[4] = (_Float16)xr3.x; p1[5] = (_Float16)xr3.y;
            p1[6] = (_Float16)xr3.z; p1[7] = (_Float16)xr3.w;
            *(v8h*)&sm.sX[nxt][xm * Iv + xs]     = p0;
            *(v8h*)&sm.sX[nxt][xm * Iv + xs + 8] = p1;
        }

        // ---- LSTM cell update: fully in registers ----
        // acc[0..3][r] = i,f,g,o pre-activations at (row hi*8+r, col cc)
#pragma unroll
        for (int r = 0; r < 8; ++r) {
            float ii = sigm_fast(acc[0][r]);
            float ff = sigm_fast(acc[1][r]);
            float gg = tanh_fast(acc[2][r]);
            float oo = sigm_fast(acc[3][r]);
            float cs = ff * creg[r] + ii * gg;
            creg[r] = cs;
            float hv = oo * tanh_fast(cs);
            sm.sH[nxt][(hi * 8 + r) * Hv + cc] = (_Float16)hv;
        }
        __syncthreads();
        cur = nxt;
    }

    // ---- Output projection: out = h_last @ W_out^T + b_out (tiny) ----
    const _Float16* hf = sm.sH[cur];
    for (int idx = tid; idx < BT * Ov; idx += NT) {
        int m = idx / Ov;
        int o = idx - m * Ov;
        float s = b_out[o];
#pragma unroll 8
        for (int k = 0; k < Hv; ++k)
            s += (float)hf[m * Hv + k] * W_out[o * Hv + k];
        out[(size_t)(b0 + m) * Ov + o] = s;
    }
}

extern "C" void kernel_launch(void* const* d_in, const int* in_sizes, int n_in,
                              void* d_out, int out_size, void* d_ws, size_t ws_size,
                              hipStream_t stream) {
    const float* x     = (const float*)d_in[0];
    const float* W_ih  = (const float*)d_in[1];
    const float* W_hh  = (const float*)d_in[2];
    const float* b_ih  = (const float*)d_in[3];
    const float* b_hh  = (const float*)d_in[4];
    const float* W_out = (const float*)d_in[5];
    const float* b_out = (const float*)d_in[6];
    float* out = (float*)d_out;

    dim3 grid(Bv / BT);   // 256 workgroups
    dim3 block(NT);       // 4 wave32s
    lstm_fused_kernel<<<grid, block, 0, stream>>>(x, W_ih, W_hh, b_ih, b_hh,
                                                  W_out, b_out, out);
}
